// Encoder_70824010711407
// MI455X (gfx1250) — matrix operand order repaired and checked by reading, not compiled, not run on hardware
//
#include <hip/hip_runtime.h>
#include <cstdint>
#include <cstddef>

typedef __attribute__((ext_vector_type(16))) _Float16 v16h;
typedef __attribute__((ext_vector_type(8)))  _Float16 v8h;
typedef __attribute__((ext_vector_type(8)))  float    v8f;

#define HC 256   // H*C
#define CH 64    // C
#define NH 4     // heads
#define NEG_SLOPE 0.2f

// ---------------------------------------------------------------- utilities

__global__ void fill_f32(float* __restrict__ p, float v, int n) {
  int i = blockIdx.x * blockDim.x + threadIdx.x;
  if (i < n) p[i] = v;
}

__global__ void cvt_f16_kernel(const float* __restrict__ in, _Float16* __restrict__ out, int n) {
  int i = blockIdx.x * blockDim.x + threadIdx.x;
  if (i < n) out[i] = (_Float16)in[i];
}

__device__ inline void atomicMaxF(float* addr, float val) {
  // sign-split float max; target initialized to -inf
  if (val >= 0.0f) atomicMax((int*)addr, __float_as_int(val));
  else             atomicMin((unsigned int*)addr, __float_as_uint(val));
}

// ------------------------------------------------- self-loop attr precompute

__global__ void degree_kernel(const int* __restrict__ dst, const float* __restrict__ ea,
                              float* __restrict__ cnt, float* __restrict__ lsum, int nE) {
  int e = blockIdx.x * blockDim.x + threadIdx.x;
  if (e < nE) {
    int d = dst[e];
    atomicAdd(&cnt[d], 1.0f);
    atomicAdd(&lsum[d], ea[e]);
  }
}

__global__ void loopattr_kernel(float* __restrict__ lattr, const float* __restrict__ cnt, int nN) {
  int i = blockIdx.x * blockDim.x + threadIdx.x;
  if (i < nN) lattr[i] = lattr[i] / fmaxf(cnt[i], 1.0f);
}

// -------------------------------------------------------- weight pre-packing
// Wpk layout (halves): [ct:32][kb:K/32][lane:32][j:16]
//   ct 0..15  -> Wl column tiles, ct 16..31 -> Wr column tiles
//   B fragment for v_wmma_f32_16x16x32_f16:
//     lanes 0-15 hold K = kb*32 + {0..7,16..23} (col = lane), lanes 16-31 hold K = +{8..15,24..31}

__global__ void pack_w_kernel(const float* __restrict__ Wl, const float* __restrict__ Wr,
                              _Float16* __restrict__ Wpk, int K) {
  int kb_n = K >> 5;
  int total = 32 * kb_n * 512;
  int idx = blockIdx.x * blockDim.x + threadIdx.x;
  if (idx >= total) return;
  int j    = idx & 15;
  int lane = (idx >> 4) & 31;
  int kb   = (idx >> 9) % kb_n;
  int ct   = idx / (kb_n * 512);
  const float* W = (ct < 16) ? Wl : Wr;
  int col = (ct & 15) * 16 + (lane & 15);
  int k   = kb * 32 + (lane >> 4) * 8 + (j & 7) + (j >> 3) * 16;
  Wpk[idx] = (_Float16)W[(size_t)k * HC + col];
}

// ---------------------------------------------------------------- WMMA GEMM
// xl = A@Wl + bl, xr = A@Wr + br fused. A16: Nrows x K f16 row-major.
// grid.x = Nrows/16 (Nrows must be a multiple of 16; N=50000 is).
// block = 256 threads = 8 waves; wave w owns column tiles [4w, 4w+4):
// waves 0-3 -> xl cols 0..255, waves 4-7 -> xr cols 0..255.
// One A fragment per K-step feeds 4 independent WMMAs (4 accumulators).

__global__ void gemm_dual_wmma(const _Float16* __restrict__ A16,
                               const _Float16* __restrict__ Wpk,
                               const float* __restrict__ bl, const float* __restrict__ br,
                               float* __restrict__ xl, float* __restrict__ xr,
                               int K) {
  int wave = threadIdx.x >> 5;
  int lane = threadIdx.x & 31;
  int half = lane >> 4;        // 0: lanes 0-15, 1: lanes 16-31
  int l16  = lane & 15;
  int ctB  = wave * 4;         // 4 column tiles per wave, never straddles Wl/Wr

  const float* bias = (ctB < 16) ? bl : br;
  float*       Out  = (ctB < 16) ? xl : xr;

  int tm = blockIdx.x * 16;
  int kb_n = K >> 5;
  // A fragment rows: lane l16 -> row tm+l16; half 0 ks {0..7,16..23}, half 1 {8..15,24..31}
  const _Float16* arow = A16 + (size_t)(tm + l16) * K + half * 8;
  const _Float16* wb0  = Wpk + ((size_t)ctB * kb_n * 32 + (size_t)lane) * 16;
  const size_t tileStride = (size_t)kb_n * 512;   // halves per column tile

  v8f acc[4] = {{}, {}, {}, {}};
  for (int kb = 0; kb < kb_n; ++kb) {
    v8h a0 = *(const v8h*)(arow + kb * 32);        // K = kb*32 + half*8 + 0..7
    v8h a1 = *(const v8h*)(arow + kb * 32 + 16);   // K = kb*32 + half*8 + 16..23
    v16h af;
#pragma unroll
    for (int j = 0; j < 8; ++j) { af[j] = a0[j]; af[8 + j] = a1[j]; }

    const _Float16* wb = wb0 + (size_t)kb * 512;
#pragma unroll
    for (int i = 0; i < 4; ++i) {
      v16h bf = *(const v16h*)(wb + (size_t)i * tileStride);   // 32B contiguous per lane
      acc[i] = __builtin_amdgcn_wmma_f32_16x16x32_f16(
          /*neg_a=*/false, af, /*neg_b=*/false, bf,
          /*c_mod=*/(short)0, acc[i], /*reuse_a=*/false, /*reuse_b=*/false);
    }
  }

  // D layout: col = tile_col + (lane&15); VGPR r -> row M = r + 8*half
#pragma unroll
  for (int i = 0; i < 4; ++i) {
    int col = ((ctB + i) & 15) * 16 + l16;
    float bv = bias[col];
#pragma unroll
    for (int r = 0; r < 8; ++r) {
      Out[(size_t)(tm + half * 8 + r) * HC + col] = acc[i][r] + bv;
    }
  }
}

// ------------------------------------------------------------- edge kernels
// Edge e < nE: (src[e], dst[e], ea[e]); e >= nE: self loop i=e-nE, attr lattr[i].
// One wave per edge; lane covers channels c = lane + 32*j, j=0..7; head = j>>1.

__global__ void edge_score_kernel(const float* __restrict__ xl, const float* __restrict__ xr,
                                  const int* __restrict__ src, const int* __restrict__ dst,
                                  const float* __restrict__ ea, const float* __restrict__ lattr,
                                  const float* __restrict__ We, const float* __restrict__ att,
                                  float* __restrict__ score, float* __restrict__ mx,
                                  int nE, int nN) {
  int e = blockIdx.x * (blockDim.x >> 5) + (threadIdx.x >> 5);
  int lane = threadIdx.x & 31;
  int nTot = nE + nN;
  if (e >= nTot) return;

  int s, d; float eav;
  if (e < nE) { s = src[e]; d = dst[e]; eav = ea[e]; }
  else        { s = d = e - nE;         eav = lattr[s]; }

  const float* xls = xl + (size_t)s * HC;
  const float* xrd = xr + (size_t)d * HC;

  float ph[NH] = {0.f, 0.f, 0.f, 0.f};
#pragma unroll
  for (int j = 0; j < 8; ++j) {
    int c = lane + 32 * j;
    float m = xls[c] + xrd[c] + eav * We[c];
    m = (m >= 0.0f) ? m : NEG_SLOPE * m;
    ph[j >> 1] += m * att[c];           // att flat (H,C): index == c
  }
#pragma unroll
  for (int off = 16; off > 0; off >>= 1) {
#pragma unroll
    for (int h = 0; h < NH; ++h) ph[h] += __shfl_xor(ph[h], off, 32);
  }
  if (lane < NH) {
    score[(size_t)e * NH + lane] = ph[lane];
    atomicMaxF(&mx[(size_t)d * NH + lane], ph[lane]);
  }
}

__global__ void edge_exp_kernel(float* __restrict__ score_ex,   // in: score, out: ex (in place)
                                const float* __restrict__ mx,
                                const int* __restrict__ dst,
                                float* __restrict__ den, int nE, int nN) {
  int idx = blockIdx.x * blockDim.x + threadIdx.x;
  int nTot = (nE + nN) * NH;
  if (idx >= nTot) return;
  int e = idx >> 2, h = idx & 3;
  int d = (e < nE) ? dst[e] : (e - nE);
  float m = mx[(size_t)d * NH + h];
  if (!(m >= -3.0e38f && m <= 3.0e38f)) m = 0.0f;   // where(isfinite(mx), mx, 0)
  float v = expf(score_ex[idx] - m);
  score_ex[idx] = v;
  atomicAdd(&den[(size_t)d * NH + h], v);
}

__global__ void edge_aggregate_kernel(const float* __restrict__ xl,
                                      const int* __restrict__ src, const int* __restrict__ dst,
                                      const float* __restrict__ ex, const float* __restrict__ den,
                                      float* __restrict__ accum, int nE, int nN) {
  int e = blockIdx.x * (blockDim.x >> 5) + (threadIdx.x >> 5);
  int lane = threadIdx.x & 31;
  int nTot = nE + nN;
  if (e >= nTot) return;

  int s, d;
  if (e < nE) { s = src[e]; d = dst[e]; }
  else        { s = d = e - nE; }

  float alpha[NH];
#pragma unroll
  for (int h = 0; h < NH; ++h)
    alpha[h] = ex[(size_t)e * NH + h] / (den[(size_t)d * NH + h] + 1e-16f);

  const float* xls = xl + (size_t)s * HC;
  float* acc = accum + (size_t)d * HC;
#pragma unroll
  for (int j = 0; j < 8; ++j) {
    int c = lane + 32 * j;
    atomicAdd(&acc[c], alpha[j >> 1] * xls[c]);
  }
}

__global__ void finalize_kernel(const float* __restrict__ accum, const float* __restrict__ bias,
                                const float* __restrict__ prelu, float* __restrict__ out,
                                int nN, int apply_prelu) {
  int idx = blockIdx.x * blockDim.x + threadIdx.x;
  if (idx >= nN * CH) return;
  int n = idx >> 6, c = idx & 63;
  const float* a = accum + (size_t)n * HC;
  float v = 0.25f * (a[c] + a[CH + c] + a[2 * CH + c] + a[3 * CH + c]) + bias[c];
  if (apply_prelu) v = (v >= 0.0f) ? v : prelu[c] * v;
  out[idx] = v;
}

// ------------------------------------------------------------------- driver

extern "C" void kernel_launch(void* const* d_in, const int* in_sizes, int n_in,
                              void* d_out, int out_size, void* d_ws, size_t ws_size,
                              hipStream_t stream) {
  const float* x     = (const float*)d_in[0];
  const int*   ei    = (const int*)d_in[1];
  const float* eattr = (const float*)d_in[2];
  const float* Wl[3]   = {(const float*)d_in[3],  (const float*)d_in[10], (const float*)d_in[17]};
  const float* bl[3]   = {(const float*)d_in[4],  (const float*)d_in[11], (const float*)d_in[18]};
  const float* Wr[3]   = {(const float*)d_in[5],  (const float*)d_in[12], (const float*)d_in[19]};
  const float* br[3]   = {(const float*)d_in[6],  (const float*)d_in[13], (const float*)d_in[20]};
  const float* We[3]   = {(const float*)d_in[7],  (const float*)d_in[14], (const float*)d_in[21]};
  const float* att[3]  = {(const float*)d_in[8],  (const float*)d_in[15], (const float*)d_in[22]};
  const float* bias[3] = {(const float*)d_in[9],  (const float*)d_in[16], (const float*)d_in[23]};
  const float* prelu   = (const float*)d_in[24];

  const int nN = in_sizes[0] / HC;     // 50000 (IN==256==HC); multiple of 16
  const int nE = in_sizes[2];          // 800000
  const int E2 = nE + nN;              // edges + self loops
  const int* src = ei;
  const int* dst = ei + nE;

  // workspace carve-up (floats; every region size is a multiple of 8 floats =
  // 32 bytes, so the f16 fragment buffers stay 32B-aligned)
  float* ws = (float*)d_ws;
  size_t o = 0;
  float* xl    = ws + o; o += (size_t)nN * HC;    // source transform
  float* xr    = ws + o; o += (size_t)nN * HC;    // dest transform; reused as accum
  float* score = ws + o; o += (size_t)E2 * NH;    // scores; reused in-place as ex
  float* mx    = ws + o; o += (size_t)nN * NH;
  float* den   = ws + o; o += (size_t)nN * NH;
  float* h1    = ws + o; o += (size_t)nN * CH;
  float* h2    = ws + o; o += (size_t)nN * CH;
  float* cnt   = ws + o; o += (size_t)nN;
  float* lattr = ws + o; o += (size_t)nN;
  _Float16* A16 = (_Float16*)(ws + o); o += (size_t)nN * HC / 2;  // N x K halves (K<=256)
  _Float16* Wpk = (_Float16*)(ws + o); o += 32 * 8 * 512 / 2;     // packed W fragments
  (void)ws_size; (void)n_in; (void)out_size;

  const int T = 256;
  auto blk = [](int n, int t) { return (n + t - 1) / t; };

  // ---- self-loop attr = segment_mean(edge_attr by dst) ----
  fill_f32<<<blk(nN, T), T, 0, stream>>>(cnt, 0.0f, nN);
  fill_f32<<<blk(nN, T), T, 0, stream>>>(lattr, 0.0f, nN);
  degree_kernel<<<blk(nE, T), T, 0, stream>>>(dst, eattr, cnt, lattr, nE);
  loopattr_kernel<<<blk(nN, T), T, 0, stream>>>(lattr, cnt, nN);

  const float* layer_in = x;
  int K = HC;                          // 256 for layer 1, 64 after
  for (int l = 0; l < 3; ++l) {
    float* layer_out = (l == 0) ? h1 : (l == 1) ? h2 : (float*)d_out;

    // segment-softmax state
    fill_f32<<<blk(nN * NH, T), T, 0, stream>>>(mx, -__builtin_huge_valf(), nN * NH);
    fill_f32<<<blk(nN * NH, T), T, 0, stream>>>(den, 0.0f, nN * NH);

    // f16 staging: activations once, weights fragment-packed once
    cvt_f16_kernel<<<blk(nN * K, T), T, 0, stream>>>(layer_in, A16, nN * K);
    pack_w_kernel<<<blk(32 * (K >> 5) * 512, T), T, 0, stream>>>(Wl[l], Wr[l], Wpk, K);

    // xl = in@Wl + bl ; xr = in@Wr + br   (WMMA f16->f32, 4 tiles per wave)
    gemm_dual_wmma<<<dim3(nN / 16), T, 0, stream>>>(A16, Wpk, bl[l], br[l], xl, xr, K);

    // per-edge attention scores + segment max
    edge_score_kernel<<<blk(E2, T / 32), T, 0, stream>>>(
        xl, xr, src, dst, eattr, lattr, We[l], att[l], score, mx, nE, nN);

    // ex = exp(score - mx[dst]); den = segment_sum(ex)
    edge_exp_kernel<<<blk(E2 * NH, T), T, 0, stream>>>(score, mx, dst, den, nE, nN);

    // accum (aliases xr, no longer needed) = segment_sum(alpha * xl[src])
    fill_f32<<<blk(nN * HC, T), T, 0, stream>>>(xr, 0.0f, nN * HC);
    edge_aggregate_kernel<<<blk(E2, T / 32), T, 0, stream>>>(
        xl, src, dst, score, den, xr, nE, nN);

    // head mean + bias (+ PReLU on final layer)
    finalize_kernel<<<blk(nN * CH, T), T, 0, stream>>>(
        xr, bias[l], prelu, layer_out, nN, (l == 2) ? 1 : 0);

    layer_in = layer_out;
    K = CH;
  }
}
